// DeepseekV3NaiveMoe_88630945120716
// MI455X (gfx1250) — matrix operand order repaired and checked by reading, not compiled, not run on hardware
//
#include <hip/hip_runtime.h>
#include <hip/hip_bf16.h>

typedef __bf16 bf16;
typedef __attribute__((ext_vector_type(16))) __bf16 v16bf;
typedef __attribute__((ext_vector_type(8)))  __bf16 v8bf;
typedef __attribute__((ext_vector_type(4)))  __bf16 v4bf;
typedef __attribute__((ext_vector_type(8)))  float  v8f;
typedef __attribute__((ext_vector_type(4)))  float  f32x4;

#define T_TOK   8192
#define H_DIM   1024
#define I_DIM   1408
#define E_NUM   32
#define K_TOP   8
#define NPAIR   (T_TOK * K_TOP)      // 65536
#define TILE_M  128
#define TILE_N  128
#define KCHUNK  32
#define LDB     40                   // LDS row stride in bf16 (80B: 16B-aligned, bank-spread)
#define WLMAX   544                  // sum ceil(cnt_e/128) <= 512 + 32
#define HBLOCKS 256                  // 256 pairs per hist/scatter block

// ---------------------------------------------------------------- routing ---

__global__ __launch_bounds__(256) void moe_hist(const int* __restrict__ topk,
                                                int* __restrict__ bh) {
  __shared__ int h[E_NUM];
  int tid = threadIdx.x;
  if (tid < E_NUM) h[tid] = 0;
  __syncthreads();
  int pair = blockIdx.x * 256 + tid;
  atomicAdd(&h[topk[pair]], 1);
  __syncthreads();
  if (tid < E_NUM) bh[blockIdx.x * E_NUM + tid] = h[tid];
}

__global__ __launch_bounds__(256) void moe_scan(int* __restrict__ bh,
                                                int* __restrict__ eoff,
                                                int* __restrict__ wlist,
                                                int* __restrict__ ntiles,
                                                int* __restrict__ zbuf) {
  __shared__ int totals[E_NUM];
  __shared__ int off[E_NUM + 1];
  __shared__ int wloff[E_NUM + 1];
  int tid = threadIdx.x;
  for (int i = tid; i < 1024; i += 256) zbuf[i] = 0;   // 4KB zero source
  if (tid < E_NUM) {
    int run = 0;
    for (int b = 0; b < HBLOCKS; ++b) {
      int v = bh[b * E_NUM + tid];
      bh[b * E_NUM + tid] = run;     // per-block prefix within expert
      run += v;
    }
    totals[tid] = run;
  }
  __syncthreads();
  if (tid == 0) {
    int s = 0, w = 0;
    for (int e = 0; e < E_NUM; ++e) {
      off[e] = s;   s += totals[e];
      wloff[e] = w; w += (totals[e] + TILE_M - 1) / TILE_M;
    }
    off[E_NUM] = s;
    wloff[E_NUM] = w;
    *ntiles = w;
  }
  __syncthreads();
  if (tid < E_NUM) {
    for (int b = 0; b < HBLOCKS; ++b) bh[b * E_NUM + tid] += off[tid];
    eoff[tid] = off[tid];
    if (tid == 0) eoff[E_NUM] = off[E_NUM];
    int base = wloff[tid];
    int n = (totals[tid] + TILE_M - 1) / TILE_M;
    for (int i = 0; i < n; ++i) wlist[base + i] = (tid << 16) | i;
  }
}

__global__ __launch_bounds__(256) void moe_scatter(const int* __restrict__ topk,
                                                   const int* __restrict__ bbase,
                                                   int* __restrict__ perm,
                                                   int* __restrict__ invpos) {
  __shared__ int eid[256];
  int tid = threadIdx.x;
  int pair = blockIdx.x * 256 + tid;
  int e = topk[pair];
  eid[tid] = e;
  __syncthreads();
  int rank = 0;
  for (int j = 0; j < tid; ++j) rank += (eid[j] == e) ? 1 : 0;   // stable rank
  int pos = bbase[blockIdx.x * E_NUM + e] + rank;
  perm[pos] = pair;
  invpos[pair] = pos;
}

// ----------------------------------------------------- precision conversion -

// hidden_states f32 -> bf16, straight copy
__global__ __launch_bounds__(256) void moe_cvt_hs(const float* __restrict__ s,
                                                  bf16* __restrict__ d) {
  size_t i = ((size_t)blockIdx.x * 256 + threadIdx.x) * 4;
  f32x4 f = *(const f32x4*)(s + i);
  v4bf o;
#pragma unroll
  for (int j = 0; j < 4; ++j) o[j] = (bf16)f[j];
  *(v4bf*)(d + i) = o;
}

// weights: src [E][K][N] f32  ->  dst [E][N][K] bf16 (transpose, 32x32 LDS tiles)
__global__ __launch_bounds__(256) void moe_cvt_w(const float* __restrict__ src,
                                                 bf16* __restrict__ dst,
                                                 int K, int N) {
  __shared__ float tile[32][33];
  int e = blockIdx.z;
  int n0 = blockIdx.x * 32, k0 = blockIdx.y * 32;
  int tid = threadIdx.x;
  int c = tid & 31, r8 = tid >> 5;
  const float* s = src + (size_t)e * K * N;
  bf16* d = dst + (size_t)e * K * N;
#pragma unroll
  for (int i = 0; i < 4; ++i) {
    int r = r8 + i * 8;
    tile[r][c] = s[(size_t)(k0 + r) * N + n0 + c];
  }
  __syncthreads();
#pragma unroll
  for (int i = 0; i < 4; ++i) {
    int n = r8 + i * 8;
    d[(size_t)(n0 + n) * K + k0 + c] = (bf16)tile[c][n];
  }
}

// ---------------------------------------------------------------- wmma utils -

union V16U { v16bf v; v8bf h[2]; };

// A fragment: lane m = lane&15, halves at K = hi*8 and 16+hi*8 (ISA 16-bit A 16x32)
__device__ __forceinline__ v16bf frag_a(const bf16* s, int row, int hi) {
  const bf16* p = s + row * LDB + hi * 8;
  V16U r;
  r.h[0] = *(const v8bf*)p;
  r.h[1] = *(const v8bf*)(p + 16);
  return r.v;
}
// B fragment: lane col = lane&15, halves at K = hi*8 and 16+hi*8 (B column-major)
__device__ __forceinline__ v16bf frag_b(const bf16* s, int col, int hi) {
  const bf16* p = s + col * LDB + hi * 8;
  V16U r;
  r.h[0] = *(const v8bf*)p;
  r.h[1] = *(const v8bf*)(p + 16);
  return r.v;
}
__device__ __forceinline__ v8f wmma_bf16(v16bf a, v16bf b, v8f c) {
  return __builtin_amdgcn_wmma_f32_16x16x32_bf16(false, a, false, b, (short)0, c,
                                                 false, false);
}

// LDS byte offset of a __shared__ object (flat LDS addr carries offset in [31:0])
__device__ __forceinline__ unsigned lds_off(const void* p) {
  return (unsigned)(uintptr_t)p;
}
// async global->LDS copy of 32 bytes (two b128 lane copies), tracked by ASYNCcnt
__device__ __forceinline__ void async_cp32(unsigned ldso, const bf16* g) {
  asm volatile("global_load_async_to_lds_b128 %0, %1, off"
               :: "v"(ldso), "v"(g) : "memory");
  asm volatile("global_load_async_to_lds_b128 %0, %1, off"
               :: "v"(ldso + 16u), "v"(g + 8) : "memory");
}
__device__ __forceinline__ void wait_async0() {
  asm volatile("s_wait_asynccnt 0x0" ::: "memory");
}

// ---------------------------------------------------------------- GEMM 1 -----
// grouped rows (via perm) x {wgT,wuT}  ->  act = silu(gate)*up   (bf16)
// all operands bf16; staging via double-buffered async copies to LDS

__global__ __launch_bounds__(256) void moe_gemm1(
    const bf16* __restrict__ hsb, const bf16* __restrict__ wgT,
    const bf16* __restrict__ wuT, const int* __restrict__ perm,
    const int* __restrict__ eoff, const int* __restrict__ wlist,
    const int* __restrict__ ntiles, const bf16* __restrict__ zbuf,
    bf16* __restrict__ act) {
  if ((int)blockIdx.y >= *ntiles) return;
  int wl = wlist[blockIdx.y];
  int e = wl >> 16, mt = wl & 0xffff;
  int off = eoff[e];
  int cnt = eoff[e + 1] - off;
  int m0 = mt * TILE_M;
  int n0 = blockIdx.x * TILE_N;

  __shared__ bf16 sA[2][TILE_M * LDB];
  __shared__ bf16 sBg[2][TILE_N * LDB];
  __shared__ bf16 sBu[2][TILE_N * LDB];

  int tid = threadIdx.x;
  int lane = tid & 31, wv = tid >> 5;
  int wm = wv >> 1, wn = wv & 1;          // 4x2 wave grid -> 32x64 per wave
  int lm = lane & 15, hi = lane >> 4;

  const v8f vz = {0.f, 0.f, 0.f, 0.f, 0.f, 0.f, 0.f, 0.f};
  v8f accG[2][4], accU[2][4];
#pragma unroll
  for (int i = 0; i < 2; ++i)
#pragma unroll
    for (int j = 0; j < 4; ++j) { accG[i][j] = vz; accU[i][j] = vz; }

  // staging decomposition: 2 threads per row, 16 bf16 (32B) each
  int rrow = tid >> 1, half = tid & 1;
  const bf16* abase;
  {
    int gr = m0 + rrow;
    abase = (gr < cnt) ? hsb + (size_t)(perm[off + gr] >> 3) * H_DIM + half * 16
                       : zbuf + half * 16;           // zbuf covers kc range
  }
  const bf16* gbase = wgT + (size_t)e * I_DIM * H_DIM + (size_t)(n0 + rrow) * H_DIM + half * 16;
  const bf16* ubase = wuT + (size_t)e * I_DIM * H_DIM + (size_t)(n0 + rrow) * H_DIM + half * 16;
  unsigned aoff0 = lds_off(&sA[0][rrow * LDB + half * 16]);
  unsigned aoff1 = lds_off(&sA[1][rrow * LDB + half * 16]);
  unsigned goff0 = lds_off(&sBg[0][rrow * LDB + half * 16]);
  unsigned goff1 = lds_off(&sBg[1][rrow * LDB + half * 16]);
  unsigned uoff0 = lds_off(&sBu[0][rrow * LDB + half * 16]);
  unsigned uoff1 = lds_off(&sBu[1][rrow * LDB + half * 16]);

  auto issue = [&](int b, int kc) {
    async_cp32(b ? aoff1 : aoff0, abase + kc);
    async_cp32(b ? goff1 : goff0, gbase + kc);
    async_cp32(b ? uoff1 : uoff0, ubase + kc);
  };

  issue(0, 0);
  int pb = 0;
  for (int kc = 0; kc < H_DIM; kc += KCHUNK, pb ^= 1) {
    wait_async0();
    __syncthreads();                       // copies into pb visible to all waves
    if (kc + KCHUNK < H_DIM) issue(pb ^ 1, kc + KCHUNK);
    const bf16* cA = sA[pb];
    const bf16* cG = sBg[pb];
    const bf16* cU = sBu[pb];
    v16bf a0 = frag_a(cA, wm * 32 + lm, hi);
    v16bf a1 = frag_a(cA, wm * 32 + 16 + lm, hi);
#pragma unroll
    for (int tn = 0; tn < 4; ++tn) {
      int col = wn * 64 + tn * 16 + lm;
      v16bf bg = frag_b(cG, col, hi);
      v16bf bu = frag_b(cU, col, hi);
      accG[0][tn] = wmma_bf16(a0, bg, accG[0][tn]);
      accG[1][tn] = wmma_bf16(a1, bg, accG[1][tn]);
      accU[0][tn] = wmma_bf16(a0, bu, accU[0][tn]);
      accU[1][tn] = wmma_bf16(a1, bu, accU[1][tn]);
    }
  }

  // epilogue: act = silu(gate)*up  (C/D layout: VGPR r -> rows r / r+8)
#pragma unroll
  for (int tm = 0; tm < 2; ++tm) {
#pragma unroll
    for (int tn = 0; tn < 4; ++tn) {
      int rbase = m0 + wm * 32 + tm * 16 + hi * 8;
      int col = n0 + wn * 64 + tn * 16 + lm;
#pragma unroll
      for (int r = 0; r < 8; ++r) {
        int row = rbase + r;
        if (row < cnt) {
          float g = accG[tm][tn][r];
          float u = accU[tm][tn][r];
          float sv = g / (1.0f + __expf(-g));
          act[(size_t)(off + row) * I_DIM + col] = (bf16)(sv * u);
        }
      }
    }
  }
}

// ---------------------------------------------------------------- GEMM 2 -----
// down = act @ wdT[e]   (f32 out, grouped rows)

__global__ __launch_bounds__(256) void moe_gemm2(
    const bf16* __restrict__ act, const bf16* __restrict__ wdT,
    const int* __restrict__ eoff, const int* __restrict__ wlist,
    const int* __restrict__ ntiles, const bf16* __restrict__ zbuf,
    float* __restrict__ down) {
  if ((int)blockIdx.y >= *ntiles) return;
  int wl = wlist[blockIdx.y];
  int e = wl >> 16, mt = wl & 0xffff;
  int off = eoff[e];
  int cnt = eoff[e + 1] - off;
  int m0 = mt * TILE_M;
  int n0 = blockIdx.x * TILE_N;

  __shared__ bf16 sA[2][TILE_M * LDB];
  __shared__ bf16 sB[2][TILE_N * LDB];

  int tid = threadIdx.x;
  int lane = tid & 31, wv = tid >> 5;
  int wm = wv >> 1, wn = wv & 1;
  int lm = lane & 15, hi = lane >> 4;

  const v8f vz = {0.f, 0.f, 0.f, 0.f, 0.f, 0.f, 0.f, 0.f};
  v8f acc[2][4];
#pragma unroll
  for (int i = 0; i < 2; ++i)
#pragma unroll
    for (int j = 0; j < 4; ++j) acc[i][j] = vz;

  int rrow = tid >> 1, half = tid & 1;
  const bf16* abase;
  {
    int gr = m0 + rrow;
    abase = (gr < cnt) ? act + (size_t)(off + gr) * I_DIM + half * 16
                       : zbuf + half * 16;
  }
  const bf16* bbase = wdT + (size_t)e * I_DIM * H_DIM + (size_t)(n0 + rrow) * I_DIM + half * 16;
  unsigned aoff0 = lds_off(&sA[0][rrow * LDB + half * 16]);
  unsigned aoff1 = lds_off(&sA[1][rrow * LDB + half * 16]);
  unsigned boff0 = lds_off(&sB[0][rrow * LDB + half * 16]);
  unsigned boff1 = lds_off(&sB[1][rrow * LDB + half * 16]);

  auto issue = [&](int b, int kc) {
    async_cp32(b ? aoff1 : aoff0, abase + kc);
    async_cp32(b ? boff1 : boff0, bbase + kc);
  };

  issue(0, 0);
  int pb = 0;
  for (int kc = 0; kc < I_DIM; kc += KCHUNK, pb ^= 1) {
    wait_async0();
    __syncthreads();
    if (kc + KCHUNK < I_DIM) issue(pb ^ 1, kc + KCHUNK);
    const bf16* cA = sA[pb];
    const bf16* cB = sB[pb];
    v16bf a0 = frag_a(cA, wm * 32 + lm, hi);
    v16bf a1 = frag_a(cA, wm * 32 + 16 + lm, hi);
#pragma unroll
    for (int tn = 0; tn < 4; ++tn) {
      v16bf b = frag_b(cB, wn * 64 + tn * 16 + lm, hi);
      acc[0][tn] = wmma_bf16(a0, b, acc[0][tn]);
      acc[1][tn] = wmma_bf16(a1, b, acc[1][tn]);
    }
  }

#pragma unroll
  for (int tm = 0; tm < 2; ++tm) {
#pragma unroll
    for (int tn = 0; tn < 4; ++tn) {
      int rbase = m0 + wm * 32 + tm * 16 + hi * 8;
      int col = n0 + wn * 64 + tn * 16 + lm;
#pragma unroll
      for (int r = 0; r < 8; ++r) {
        int row = rbase + r;
        if (row < cnt)
          down[(size_t)(off + row) * H_DIM + col] = acc[tm][tn][r];
      }
    }
  }
}

// ---------------------------------------------------------------- combine ----

__global__ __launch_bounds__(256) void moe_combine(const float* __restrict__ down,
                                                   const int* __restrict__ invpos,
                                                   const float* __restrict__ wts,
                                                   float* __restrict__ out) {
  __shared__ int pos[K_TOP];
  __shared__ float wk[K_TOP];
  int t = blockIdx.x;
  int tid = threadIdx.x;
  if (tid < K_TOP) {
    pos[tid] = invpos[t * K_TOP + tid];
    wk[tid] = wts[t * K_TOP + tid];
  }
  __syncthreads();
  for (int h = tid; h < H_DIM; h += 256) {
    float a = 0.f;
#pragma unroll
    for (int k = 0; k < K_TOP; ++k)
      a += wk[k] * down[(size_t)pos[k] * H_DIM + h];
    out[(size_t)t * H_DIM + h] = a;
  }
}

// ---------------------------------------------------------------- launch -----

extern "C" void kernel_launch(void* const* d_in, const int* in_sizes, int n_in,
                              void* d_out, int out_size, void* d_ws, size_t ws_size,
                              hipStream_t stream) {
  const float* hs  = (const float*)d_in[0];
  const int* topk  = (const int*)d_in[1];
  const float* wts = (const float*)d_in[2];
  const float* wg  = (const float*)d_in[3];
  const float* wu  = (const float*)d_in[4];
  const float* wd  = (const float*)d_in[5];
  float* out = (float*)d_out;

  char* ws = (char*)d_ws;
  int* perm   = (int*)ws;                       // NPAIR
  int* invpos = perm + NPAIR;                   // NPAIR
  int* bh     = invpos + NPAIR;                 // HBLOCKS*E
  int* eoff   = bh + HBLOCKS * E_NUM;           // E+1 (padded 64)
  int* wlist  = eoff + 64;                      // WLMAX (padded 1024)
  int* ntiles = wlist + 1024;                   // 1 (padded 64)
  int* zbuf   = ntiles + 64;                    // 1024 ints = 4KB zeros

  size_t o = (size_t)1 << 20;
  bf16* act   = (bf16*)(ws + o);  o += (size_t)NPAIR * I_DIM * sizeof(bf16);
  float* down = (float*)(ws + o); o += (size_t)NPAIR * H_DIM * sizeof(float);
  bf16* hsb   = (bf16*)(ws + o);  o += (size_t)T_TOK * H_DIM * sizeof(bf16);
  bf16* wgT   = (bf16*)(ws + o);  o += (size_t)E_NUM * H_DIM * I_DIM * sizeof(bf16);
  bf16* wuT   = (bf16*)(ws + o);  o += (size_t)E_NUM * H_DIM * I_DIM * sizeof(bf16);
  bf16* wdT   = (bf16*)(ws + o);  o += (size_t)E_NUM * H_DIM * I_DIM * sizeof(bf16);

  moe_hist<<<HBLOCKS, 256, 0, stream>>>(topk, bh);
  moe_scan<<<1, 256, 0, stream>>>(bh, eoff, wlist, ntiles, zbuf);
  moe_scatter<<<HBLOCKS, 256, 0, stream>>>(topk, bh, perm, invpos);
  moe_cvt_hs<<<(T_TOK * H_DIM) / (256 * 4), 256, 0, stream>>>(hs, hsb);
  moe_cvt_w<<<dim3(I_DIM / 32, H_DIM / 32, E_NUM), 256, 0, stream>>>(wg, wgT, H_DIM, I_DIM);
  moe_cvt_w<<<dim3(I_DIM / 32, H_DIM / 32, E_NUM), 256, 0, stream>>>(wu, wuT, H_DIM, I_DIM);
  moe_cvt_w<<<dim3(H_DIM / 32, I_DIM / 32, E_NUM), 256, 0, stream>>>(wd, wdT, I_DIM, H_DIM);
  moe_gemm1<<<dim3(I_DIM / TILE_N, WLMAX), 256, 0, stream>>>(
      hsb, wgT, wuT, perm, eoff, wlist, ntiles, (const bf16*)zbuf, act);
  moe_gemm2<<<dim3(H_DIM / TILE_N, WLMAX), 256, 0, stream>>>(
      act, wdT, eoff, wlist, ntiles, (const bf16*)zbuf, down);
  moe_combine<<<T_TOK, 256, 0, stream>>>(down, invpos, wts, out);
}
